// SparseAttention_72997264163086
// MI455X (gfx1250) — matrix operand order repaired and checked by reading, not compiled, not run on hardware
//
#include <hip/hip_runtime.h>
#include <hip/hip_bf16.h>

// Problem constants (match reference)
#define DIMC  1024
#define HWC   4096
#define BB    4
#define KTOP  9
#define SCALE_F 0.03125f   // 1024^-0.5 == 1/32 exactly

typedef __attribute__((ext_vector_type(16))) _Float16 v16h;
typedef __attribute__((ext_vector_type(8)))  _Float16 v8h;
typedef __attribute__((ext_vector_type(8)))  float    v8f;

// ---------------------------------------------------------------------------
// Blocked WMMA K-loop: one wave computes a 64x64 output block as a 4x4 grid
// of 16x16 v_wmma_f32_16x16x32_f16 tiles. A row-major (M x K); B supplied as
// BT row-major (N x K) i.e. column-major B.
// Fragment addressing uses contiguous K order (k = 16*(L>>4) + s), same for A and B:
//   A 16x32: lane L -> row M=L&15; K {0..15} (L<16) or {16..31}
//   B 32x16: lane L -> col N=L&15; 16 contiguous K at half-offset (L>=16)*16
// Per K-step: 4 B frags + 4 A frags (16 x b128 loads) feed 16 WMMAs.
// ---------------------------------------------------------------------------
__device__ __forceinline__ void wmma_block64(const _Float16* __restrict__ A, int lda,
                                             const _Float16* __restrict__ BT, int ldb,
                                             int Ktot, int lane, v8f acc[4][4]) {
  const int asel = (lane >> 4) << 4;   // 0 or 16 halves
  const int bsel = (lane >> 4) << 4;   // 0 or 16 halves
  const int mr   = lane & 15;
  const _Float16* arow[4];
  const _Float16* brow[4];
#pragma unroll
  for (int i = 0; i < 4; ++i) arow[i] = A  + (size_t)(i * 16 + mr) * lda + asel;
#pragma unroll
  for (int j = 0; j < 4; ++j) brow[j] = BT + (size_t)(j * 16 + mr) * ldb + bsel;

  for (int kb = 0; kb < Ktot; kb += 32) {
    v16h bf[4];
#pragma unroll
    for (int j = 0; j < 4; ++j) bf[j] = *(const v16h*)(brow[j] + kb);
#pragma unroll
    for (int i = 0; i < 4; ++i) {
      v8h alo = *(const v8h*)(arow[i] + kb);
      v8h ahi = *(const v8h*)(arow[i] + kb + 8);
      v16h a;
#pragma unroll
      for (int t = 0; t < 8; ++t) { a[t] = alo[t]; a[t + 8] = ahi[t]; }
#pragma unroll
      for (int j = 0; j < 4; ++j)
        acc[i][j] = __builtin_amdgcn_wmma_f32_16x16x32_f16(
            false, a, false, bf[j], (short)0, acc[i][j], false, false);
    }
  }
}

// Wave/tile geometry shared by all GEMMs: 256 threads = 8 waves arranged
// 4(M) x 2(N); block tile = 256 x 128.
__device__ __forceinline__ void wave_tile(int tid, int bx, int by,
                                          int& lane, int& tileM, int& tileN) {
  lane = tid & 31;
  const int wave = tid >> 5;
  tileM = by * 256 + (wave >> 1) * 64;
  tileN = bx * 128 + (wave & 1) * 64;
}

// ---------------------------------------------------------------------------
// Conversion / staging kernels
// ---------------------------------------------------------------------------
__global__ __launch_bounds__(256) void sa_cvt_f16(const float* __restrict__ in,
                                                  _Float16* __restrict__ out, int n) {
  int i = blockIdx.x * 256 + threadIdx.x;
  if (i < n) out[i] = (_Float16)in[i];
}

// x (B, D, HW) f32  ->  xT (B, HW, D) f16
__global__ __launch_bounds__(256) void sa_transpose_cvt(const float* __restrict__ x,
                                                        _Float16* __restrict__ xt) {
  int i = blockIdx.x * 256 + threadIdx.x;           // over B*HW*D
  int e = i & (DIMC - 1);
  int rest = i >> 10;                               // b*HW + n
  int n = rest & (HWC - 1);
  int b = rest >> 12;
  xt[i] = (_Float16)x[((size_t)b * DIMC + e) * HWC + n];
}

// ---------------------------------------------------------------------------
// GEMM 1 (orientation M=n, N=d): C[n][d] = sum_e XT[n][e] * Wqkv[d][e].
// A = XT (HW x D row-major), BT = Wqkv (3D x D row-major, as given).
// Lane holds fixed d, r-run walks n -> packed v8h stores for Q,K (row-major
// D x HW); V written transposed (HW x D) with scalar stores for GEMM 3.
// Column tiles never straddle the Q/K/V boundaries (16 | 1024).
// ---------------------------------------------------------------------------
__global__ __launch_bounds__(256) void sa_qkv_gemm(const _Float16* __restrict__ W,
                                                   const _Float16* __restrict__ XT,
                                                   _Float16* __restrict__ Q,
                                                   _Float16* __restrict__ Km,
                                                   _Float16* __restrict__ VT) {
  const int b = blockIdx.z;
  int lane, tileM, tileN;
  wave_tile(threadIdx.x, blockIdx.x, blockIdx.y, lane, tileM, tileN);
  const _Float16* A  = XT + (size_t)b * HWC * DIMC + (size_t)tileM * DIMC;
  const _Float16* BT = W  + (size_t)tileN * DIMC;
  v8f acc[4][4] = {};
  wmma_block64(A, DIMC, BT, DIMC, DIMC, lane, acc);

  const int mof  = (lane >> 4) << 3;
  const int ncol = lane & 15;
#pragma unroll
  for (int j = 0; j < 4; ++j) {
    const int d = tileN + j * 16 + ncol;       // 0..3071
    if (tileN + j * 16 < DIMC) {
      _Float16* dst = Q + (size_t)b * DIMC * HWC + (size_t)d * HWC;
#pragma unroll
      for (int i = 0; i < 4; ++i) {
        v8h o;
#pragma unroll
        for (int r = 0; r < 8; ++r) o[r] = (_Float16)acc[i][j][r];
        *(v8h*)(dst + tileM + i * 16 + mof) = o;
      }
    } else if (tileN + j * 16 < 2 * DIMC) {
      _Float16* dst = Km + (size_t)b * DIMC * HWC + (size_t)(d - DIMC) * HWC;
#pragma unroll
      for (int i = 0; i < 4; ++i) {
        v8h o;
#pragma unroll
        for (int r = 0; r < 8; ++r) o[r] = (_Float16)acc[i][j][r];
        *(v8h*)(dst + tileM + i * 16 + mof) = o;
      }
    } else {
      const int dv = d - 2 * DIMC;
      _Float16* dst = VT + (size_t)b * HWC * DIMC + dv;
#pragma unroll
      for (int i = 0; i < 4; ++i)
#pragma unroll
        for (int r = 0; r < 8; ++r)
          dst[(size_t)(tileM + i * 16 + mof + r) * DIMC] = (_Float16)acc[i][j][r];
    }
  }
}

// ---------------------------------------------------------------------------
// GEMM 2 (orientation M=e, N=d): C[e][d] = sum_n K[e][n]*Q[d][n] = S[d][e].
// A = K, BT = Q (both D x HW row-major). Lane holds fixed d, r-run walks e
// -> packed v8f stores into S row-major (D x D). SCALE fused.
// ---------------------------------------------------------------------------
__global__ __launch_bounds__(256) void sa_scores_gemm(const _Float16* __restrict__ Q,
                                                      const _Float16* __restrict__ Km,
                                                      float* __restrict__ S) {
  const int b = blockIdx.z;
  int lane, tileM, tileN;
  wave_tile(threadIdx.x, blockIdx.x, blockIdx.y, lane, tileM, tileN);
  const _Float16* A  = Km + (size_t)b * DIMC * HWC + (size_t)tileM * HWC;
  const _Float16* BT = Q  + (size_t)b * DIMC * HWC + (size_t)tileN * HWC;
  v8f acc[4][4] = {};
  wmma_block64(A, HWC, BT, HWC, HWC, lane, acc);

  const int mof  = (lane >> 4) << 3;
  const int ncol = lane & 15;
#pragma unroll
  for (int j = 0; j < 4; ++j) {
    const int d = tileN + j * 16 + ncol;
    float* dst = S + (size_t)b * DIMC * DIMC + (size_t)d * DIMC;
#pragma unroll
    for (int i = 0; i < 4; ++i) {
      v8f o;
#pragma unroll
      for (int r = 0; r < 8; ++r) o[r] = acc[i][j][r] * SCALE_F;
      *(v8f*)(dst + tileM + i * 16 + mof) = o;
    }
  }
}

// ---------------------------------------------------------------------------
// Top-K(9) + diagonal sparse softmax over each row of scores (length D=1024).
// One 256-thread block (8 wave32) per row. 9 iterated max-reductions find the
// K-th largest value; keep = (score >= thresh) || diagonal; masked softmax.
// ---------------------------------------------------------------------------
__device__ __forceinline__ float sa_red_max(float v, float* red, int t) {
#pragma unroll
  for (int off = 16; off > 0; off >>= 1) v = fmaxf(v, __shfl_xor(v, off));
  if ((t & 31) == 0) red[t >> 5] = v;
  __syncthreads();
  float r = red[0];
#pragma unroll
  for (int w = 1; w < 8; ++w) r = fmaxf(r, red[w]);
  __syncthreads();
  return r;
}
__device__ __forceinline__ float sa_red_sum(float v, float* red, int t) {
#pragma unroll
  for (int off = 16; off > 0; off >>= 1) v += __shfl_xor(v, off);
  if ((t & 31) == 0) red[t >> 5] = v;
  __syncthreads();
  float r = 0.f;
#pragma unroll
  for (int w = 0; w < 8; ++w) r += red[w];
  __syncthreads();
  return r;
}

__global__ __launch_bounds__(256) void sa_topk_softmax(const float* __restrict__ S,
                                                       _Float16* __restrict__ P) {
  const int row = blockIdx.x;            // b*D + d
  const int d   = row & (DIMC - 1);
  const int t   = threadIdx.x;
  const float* Sr = S + (size_t)row * DIMC;
  __shared__ float sv[DIMC];
  __shared__ float red[8];
#pragma unroll
  for (int i = 0; i < 4; ++i) sv[t + 256 * i] = Sr[t + 256 * i];
  __syncthreads();

  float thresh = 0.f;
  for (int it = 0; it < KTOP; ++it) {
    float lm = -INFINITY;
#pragma unroll
    for (int i = 0; i < 4; ++i) lm = fmaxf(lm, sv[t + 256 * i]);
    float rm = sa_red_max(lm, red, t);
    thresh = rm;
#pragma unroll
    for (int i = 0; i < 4; ++i)
      if (sv[t + 256 * i] == rm) sv[t + 256 * i] = -INFINITY;
    __syncthreads();
  }

  float lm = -INFINITY;
#pragma unroll
  for (int i = 0; i < 4; ++i) {
    int e = t + 256 * i;
    float v = Sr[e];
    if (v >= thresh || e == d) lm = fmaxf(lm, v);
  }
  const float mx = sa_red_max(lm, red, t);

  float ls = 0.f;
#pragma unroll
  for (int i = 0; i < 4; ++i) {
    int e = t + 256 * i;
    float v = Sr[e];
    if (v >= thresh || e == d) ls += __expf(v - mx);
  }
  const float Z = sa_red_sum(ls, red, t);
  const float rz = 1.0f / Z;

  _Float16* Pr = P + (size_t)row * DIMC;
#pragma unroll
  for (int i = 0; i < 4; ++i) {
    int e = t + 256 * i;
    float v = Sr[e];
    float o = (v >= thresh || e == d) ? __expf(v - mx) * rz : 0.f;
    Pr[e] = (_Float16)o;
  }
}

// ---------------------------------------------------------------------------
// GEMM 3 (orientation M=d, N=n): O[d][n] = sum_e attn[d][e] * V[e][n].
// A = attn (D x D row-major), BT = VT (HW x D). Output stored transposed
// OT[n*D + m]: lane holds fixed n, r-run walks m -> packed v8h stores.
// ---------------------------------------------------------------------------
__global__ __launch_bounds__(256) void sa_attnv_gemm(const _Float16* __restrict__ P,
                                                     const _Float16* __restrict__ VT,
                                                     _Float16* __restrict__ OT) {
  const int b = blockIdx.z;
  int lane, tileM, tileN;
  wave_tile(threadIdx.x, blockIdx.x, blockIdx.y, lane, tileM, tileN);
  const _Float16* A  = P  + (size_t)b * DIMC * DIMC + (size_t)tileM * DIMC;
  const _Float16* BT = VT + (size_t)b * HWC * DIMC + (size_t)tileN * DIMC;
  v8f acc[4][4] = {};
  wmma_block64(A, DIMC, BT, DIMC, DIMC, lane, acc);

  const int mof  = (lane >> 4) << 3;
  const int ncol = lane & 15;
#pragma unroll
  for (int j = 0; j < 4; ++j) {
    const int n = tileN + j * 16 + ncol;
    _Float16* dst = OT + (size_t)b * HWC * DIMC + (size_t)n * DIMC;
#pragma unroll
    for (int i = 0; i < 4; ++i) {
      v8h o;
#pragma unroll
      for (int r = 0; r < 8; ++r) o[r] = (_Float16)acc[i][j][r];
      *(v8h*)(dst + tileM + i * 16 + mof) = o;
    }
  }
}

// ---------------------------------------------------------------------------
// GEMM 4 (orientation M=n, N=d): C[n][d] = sum_e OT[n][e] * Wproj[d][e].
// A = OT (HW x D), BT = Wproj (D x D, as given). Lane holds fixed d: one bias
// load per lane per j-tile; r-run walks n -> packed v8f stores to f32 output.
// ---------------------------------------------------------------------------
__global__ __launch_bounds__(256) void sa_proj_gemm(const _Float16* __restrict__ WP,
                                                    const _Float16* __restrict__ OT,
                                                    const float* __restrict__ bias,
                                                    float* __restrict__ out) {
  const int b = blockIdx.z;
  int lane, tileM, tileN;
  wave_tile(threadIdx.x, blockIdx.x, blockIdx.y, lane, tileM, tileN);
  const _Float16* A  = OT + (size_t)b * HWC * DIMC + (size_t)tileM * DIMC;
  const _Float16* BT = WP + (size_t)tileN * DIMC;
  v8f acc[4][4] = {};
  wmma_block64(A, DIMC, BT, DIMC, DIMC, lane, acc);

  const int mof  = (lane >> 4) << 3;
  const int ncol = lane & 15;
#pragma unroll
  for (int j = 0; j < 4; ++j) {
    const int d = tileN + j * 16 + ncol;
    const float bv = bias[d];
    float* dst = out + (size_t)b * DIMC * HWC + (size_t)d * HWC;
#pragma unroll
    for (int i = 0; i < 4; ++i) {
      v8f o;
#pragma unroll
      for (int r = 0; r < 8; ++r) o[r] = acc[i][j][r] + bv;
      *(v8f*)(dst + tileM + i * 16 + mof) = o;
    }
  }
}

// ---------------------------------------------------------------------------
// Host launch
// ---------------------------------------------------------------------------
extern "C" void kernel_launch(void* const* d_in, const int* in_sizes, int n_in,
                              void* d_out, int out_size, void* d_ws, size_t ws_size,
                              hipStream_t stream) {
  const float* x      = (const float*)d_in[0];   // (B, D, H, W)
  const float* w_qkv  = (const float*)d_in[1];   // (3D, D)
  const float* w_proj = (const float*)d_in[2];   // (D, D)
  const float* b_proj = (const float*)d_in[3];   // (D,)
  float* out = (float*)d_out;                    // (B, D, H, W)

  // Workspace layout (bytes)
  char* ws = (char*)d_ws;
  const size_t SZ_XT   = (size_t)BB * HWC * DIMC * 2;      // 32 MB
  const size_t SZ_WQKV = (size_t)3 * DIMC * DIMC * 2;      //  6 MB
  const size_t SZ_WPRJ = (size_t)DIMC * DIMC * 2;          //  2 MB
  const size_t SZ_QK   = (size_t)BB * DIMC * HWC * 2;      // 32 MB each
  const size_t SZ_SC   = (size_t)BB * DIMC * DIMC * 4;     // 16 MB
  const size_t SZ_AT   = (size_t)BB * DIMC * DIMC * 2;     //  8 MB

  size_t off = 0;
  _Float16* XT  = (_Float16*)(ws + off); off += SZ_XT;
  _Float16* WQ  = (_Float16*)(ws + off); off += SZ_WQKV;
  _Float16* WP  = (_Float16*)(ws + off); off += SZ_WPRJ;
  _Float16* Qh  = (_Float16*)(ws + off); off += SZ_QK;
  _Float16* Kh  = (_Float16*)(ws + off); off += SZ_QK;
  _Float16* VT  = (_Float16*)(ws + off); off += SZ_QK;
  float*    Sc  = (float*)   (ws + off); off += SZ_SC;
  _Float16* Pa  = (_Float16*)(ws + off); off += SZ_AT;
  _Float16* OT  = (_Float16*)(ws + off); off += SZ_QK;
  (void)off; (void)ws_size; (void)in_sizes; (void)n_in; (void)out_size;

  // Stage f16 operands
  sa_transpose_cvt<<<(BB * HWC * DIMC) / 256, 256, 0, stream>>>(x, XT);
  sa_cvt_f16<<<(3 * DIMC * DIMC) / 256, 256, 0, stream>>>(w_qkv, WQ, 3 * DIMC * DIMC);
  sa_cvt_f16<<<(DIMC * DIMC) / 256, 256, 0, stream>>>(w_proj, WP, DIMC * DIMC);

  // GEMM 1: QKV, M=HW (4096) x N=3D (3072), K=D
  sa_qkv_gemm<<<dim3((3 * DIMC) / 128, HWC / 256, BB), 256, 0, stream>>>(WQ, XT, Qh, Kh, VT);
  // GEMM 2: scores, M=e (1024) x N=d (1024), K=HW
  sa_scores_gemm<<<dim3(DIMC / 128, DIMC / 256, BB), 256, 0, stream>>>(Qh, Kh, Sc);
  // top-k + diagonal sparse softmax
  sa_topk_softmax<<<BB * DIMC, 256, 0, stream>>>(Sc, Pa);
  // GEMM 3: O = attn @ V, M=d (1024) x N=n (4096), K=D (stored transposed)
  sa_attnv_gemm<<<dim3(HWC / 128, DIMC / 256, BB), 256, 0, stream>>>(Pa, VT, OT);
  // GEMM 4: out = Wproj @ O + bias, M=n (4096) x N=d (1024), K=D
  sa_proj_gemm<<<dim3(DIMC / 128, HWC / 256, BB), 256, 0, stream>>>(WP, OT, b_proj, out);
}
